// RelativeMultiHeadAttention_13597866459777
// MI455X (gfx1250) — compile-verified, matrix operand order checked by reading, not run on hardware
//
#include <hip/hip_runtime.h>
#include <cstdint>

// Problem constants (match reference)
#define DM   1024      // d_model
#define NH   16        // heads
#define DH   64        // head dim
#define BB   2         // batch
#define SS   2048      // seq len
#define MTOT (BB*SS)   // 4096 rows for the projection GEMMs

typedef __attribute__((ext_vector_type(16))) __bf16 v16bf;
typedef __attribute__((ext_vector_type(8)))  __bf16 v8bf;
typedef __attribute__((ext_vector_type(4)))  __bf16 v4bf;
typedef __attribute__((ext_vector_type(8)))  float  v8f;
typedef __attribute__((ext_vector_type(4)))  unsigned int u32x4;
typedef __attribute__((ext_vector_type(8)))  int i32x8;
typedef __attribute__((ext_vector_type(4)))  int i32x4;

#if defined(__has_builtin)
# if __has_builtin(__builtin_amdgcn_tensor_load_to_lds)
#  define USE_TDM 1
# endif
#endif
#ifndef USE_TDM
# define USE_TDM 0
#endif

union Frag { v16bf v; v8bf h[2]; };

__device__ __forceinline__ v8f wmma_bf16(const Frag& a, const Frag& b, v8f c) {
  // D = A(16x32 bf16) * B(32x16 bf16) + C(16x16 f32)
  return __builtin_amdgcn_wmma_f32_16x16x32_bf16(false, a.v, false, b.v,
                                                 (short)0, c, false, false);
}

__device__ __forceinline__ v8f vzero8() { v8f z = {0.f,0.f,0.f,0.f,0.f,0.f,0.f,0.f}; return z; }

// Build a bf16 A-fragment from a fp32 LDS row (K-slices per documented layout:
// lanes 0-15 carry K 0-7 / 16-23, lanes 16-31 carry K 8-15 / 24-31).
__device__ __forceinline__ Frag frag_a_f32(const float* rowp, int half) {
  const float4 a0 = *(const float4*)(rowp + half * 8);
  const float4 a1 = *(const float4*)(rowp + half * 8 + 4);
  const float4 a2 = *(const float4*)(rowp + 16 + half * 8);
  const float4 a3 = *(const float4*)(rowp + 16 + half * 8 + 4);
  Frag f;
  f.v[0]=(__bf16)a0.x;  f.v[1]=(__bf16)a0.y;  f.v[2]=(__bf16)a0.z;  f.v[3]=(__bf16)a0.w;
  f.v[4]=(__bf16)a1.x;  f.v[5]=(__bf16)a1.y;  f.v[6]=(__bf16)a1.z;  f.v[7]=(__bf16)a1.w;
  f.v[8]=(__bf16)a2.x;  f.v[9]=(__bf16)a2.y;  f.v[10]=(__bf16)a2.z; f.v[11]=(__bf16)a2.w;
  f.v[12]=(__bf16)a3.x; f.v[13]=(__bf16)a3.y; f.v[14]=(__bf16)a3.z; f.v[15]=(__bf16)a3.w;
  return f;
}

// ---------------------------------------------------------------------------
// TDM: DMA a 128x32 fp32 tile (row stride = strideElems) from global to LDS.
// D# per CDNA5 ISA:
//   group0: count=1 | lds_addr | global_addr(57b) | type=2
//   group1: data_size=4B, pad_enable, pad every 32 DW by 4 DW (-> 144B LDS
//           row stride, bank-conflict-free b128 fragment reads),
//           tensor_dim0=strideElems, tile_dim0=32, tile_dim1=128,
//           tensor_dim0_stride=strideElems
// ---------------------------------------------------------------------------
__device__ __forceinline__ void tdm_load_tile_128x32_f32(
    unsigned lds_off, const float* gsrc, unsigned strideElems)
{
#if USE_TDM
  const unsigned long long ga = (unsigned long long)(uintptr_t)gsrc;
  u32x4 g0;
  g0[0] = 1u;                                            // count=1
  g0[1] = lds_off;                                       // lds_addr
  g0[2] = (unsigned)ga;                                  // global_addr[31:0]
  g0[3] = (unsigned)((ga >> 32) & 0x01FFFFFFull) | (2u << 30); // addr[56:32] | type=2
  i32x8 g1;
  g1[0] = (2 << 16) | (1 << 20) | (4 << 22) | (3 << 25); // 4B | pad_en | ivl=32DW | amt=4DW
  g1[1] = (int)((strideElems & 0xFFFFu) << 16);          // tensor_dim0 lo16
  g1[2] = (int)((strideElems >> 16) & 0xFFFFu);          // tensor_dim0 hi16, dim1 lo16=0
  g1[3] = (int)(0x10u | (32u << 16));                    // dim1 hi16 (1M rows) | tile_dim0=32
  g1[4] = 128;                                           // tile_dim1=128, tile_dim2=0
  g1[5] = (int)strideElems;                              // tensor_dim0_stride lo32
  g1[6] = 0;                                             // stride hi16, dim1_stride lo16
  g1[7] = 0;
  i32x4 gz = {0, 0, 0, 0};
# if __clang_major__ >= 23
  i32x8 gz8 = {0, 0, 0, 0, 0, 0, 0, 0};
  __builtin_amdgcn_tensor_load_to_lds(g0, g1, gz, gz, gz8, 0);
# else
  __builtin_amdgcn_tensor_load_to_lds(g0, g1, gz, gz, 0);
# endif
#else
  (void)lds_off; (void)gsrc; (void)strideElems;
#endif
}

__device__ __forceinline__ void wait_tensorcnt0() {
#if defined(__has_builtin) && __has_builtin(__builtin_amdgcn_s_wait_tensorcnt)
  __builtin_amdgcn_s_wait_tensorcnt(0);
#else
  asm volatile("s_wait_tensorcnt 0x0" ::: "memory");
#endif
}
__device__ __forceinline__ void wait_tensorcnt1() {
#if defined(__has_builtin) && __has_builtin(__builtin_amdgcn_s_wait_tensorcnt)
  __builtin_amdgcn_s_wait_tensorcnt(1);
#else
  asm volatile("s_wait_tensorcnt 0x1" ::: "memory");
#endif
}

// ---------------------------------------------------------------------------
// Generic GEMM: C[M,N] = A[M,K] * W[K,N] (+ bias[N]).
// A tile: TDM double-buffered fp32 in LDS (144B row stride), cvt->bf16 at
// fragment build. W tile: fp32->bf16 transpose staging. 8 waves, 128x128x32.
// ---------------------------------------------------------------------------
__global__ __launch_bounds__(256) void k_gemm_bias(
    const float* __restrict__ A, const float* __restrict__ W,
    const float* __restrict__ bias, float* __restrict__ C,
    int M, int N, int K)
{
  __shared__ float  Af[2][128][36];   // TDM destination, padded 32+4 DW rows
  __shared__ __bf16 Bl[128][48];      // [n][k] padded (W transposed on stage)
  const int t    = threadIdx.x;
  const int wave = t >> 5, lane = t & 31;
  const int half = lane >> 4, l = lane & 15;
  const int wm   = wave >> 2, wn = wave & 3;
  const int mBase = blockIdx.y * 128, nBase = blockIdx.x * 128;
  const unsigned ldsA0 = (unsigned)(uintptr_t)&Af[0][0][0];
  const unsigned ldsA1 = (unsigned)(uintptr_t)&Af[1][0][0];

  v8f acc[4][2];
#pragma unroll
  for (int mi = 0; mi < 4; ++mi)
#pragma unroll
    for (int ni = 0; ni < 2; ++ni) acc[mi][ni] = vzero8();

#if USE_TDM
  if (wave == 0)     // prologue: issue first A tile
    tdm_load_tile_128x32_f32(ldsA0, A + (size_t)mBase * K, (unsigned)K);
#endif

  for (int k0 = 0; k0 < K; k0 += 32) {
    const int buf = (k0 >> 5) & 1;
#if USE_TDM
    if (wave == 0) {                     // async: issue next tile, wait current
      if (k0 + 32 < K) {
        tdm_load_tile_128x32_f32(buf ? ldsA0 : ldsA1,
                                 A + (size_t)mBase * K + k0 + 32, (unsigned)K);
        wait_tensorcnt1();
      } else {
        wait_tensorcnt0();
      }
    }
#else
#pragma unroll
    for (int i = 0; i < 4; ++i) {        // manual fp32 A staging fallback
      int f = t + i * 256;
      int row = f >> 3, c4 = (f & 7) << 2;
      const float* ap = A + (size_t)(mBase + row) * K + k0 + c4;
      *(float4*)&Af[buf][row][c4] = *(const float4*)ap;
    }
#endif
#pragma unroll
    for (int i = 0; i < 4; ++i) {        // stage W 32x128, transpose to [n][k]
      int f = t + i * 256;
      int kk = f >> 5, n4 = (f & 31) << 2;
      const float* wp = W + (size_t)(k0 + kk) * N + nBase + n4;
      const float4 d = *(const float4*)wp;
      __builtin_prefetch(wp + (size_t)32 * N, 0, 3);
      Bl[n4+0][kk]=(__bf16)d.x; Bl[n4+1][kk]=(__bf16)d.y;
      Bl[n4+2][kk]=(__bf16)d.z; Bl[n4+3][kk]=(__bf16)d.w;
    }
    __syncthreads();
#pragma unroll
    for (int mi = 0; mi < 4; ++mi) {
      const Frag fa = frag_a_f32(&Af[buf][wm * 64 + mi * 16 + l][0], half);
#pragma unroll
      for (int ni = 0; ni < 2; ++ni) {
        Frag fb;
        const int bc = wn * 32 + ni * 16 + l;
        fb.h[0] = *(const v8bf*)&Bl[bc][half * 8];
        fb.h[1] = *(const v8bf*)&Bl[bc][16 + half * 8];
        acc[mi][ni] = wmma_bf16(fa, fb, acc[mi][ni]);
      }
    }
    __syncthreads();
  }
#pragma unroll
  for (int mi = 0; mi < 4; ++mi)
#pragma unroll
    for (int ni = 0; ni < 2; ++ni) {
      const int mrow = mBase + wm * 64 + mi * 16 + half * 8;
      const int ncol = nBase + wn * 32 + ni * 16 + l;
      const float bv = bias ? bias[ncol] : 0.f;
#pragma unroll
      for (int r = 0; r < 8; ++r)
        C[(size_t)(mrow + r) * N + ncol] = acc[mi][ni][r] + bv;
    }
}

// ---------------------------------------------------------------------------
// Score GEMM per (b,h): out[bh,m,n] = sum_d (q[b,m,h,d]+hb[h,d]) * kp[b,n,h,d]
// kp rows are N-major [s2][d] -> stage B without transpose.
// ---------------------------------------------------------------------------
__global__ __launch_bounds__(256) void k_score(
    const float* __restrict__ qp, const float* __restrict__ kp,
    const float* __restrict__ hb, float* __restrict__ out)
{
  __shared__ __bf16 Al[128][48];
  __shared__ __bf16 Bl[128][48];
  const int bh = blockIdx.z;
  const int b = bh / NH, h = bh % NH, hoff = h * DH;
  const int t = threadIdx.x;
  const int wave = t >> 5, lane = t & 31;
  const int half = lane >> 4, l = lane & 15;
  const int wm = wave >> 2, wn = wave & 3;
  const int mBase = blockIdx.y * 128, nBase = blockIdx.x * 128;

  v8f acc[4][2];
#pragma unroll
  for (int mi = 0; mi < 4; ++mi)
#pragma unroll
    for (int ni = 0; ni < 2; ++ni) acc[mi][ni] = vzero8();

#pragma unroll
  for (int k0 = 0; k0 < DH; k0 += 32) {
#pragma unroll
    for (int i = 0; i < 4; ++i) {          // A = q + head bias
      int f = t + i * 256;
      int row = f >> 3, c4 = (f & 7) << 2;
      const float4 d  = *(const float4*)(qp + (size_t)(b*SS + mBase + row)*DM + hoff + k0 + c4);
      const float4 ub = *(const float4*)(hb + hoff + k0 + c4);
      v4bf b4;
      b4[0]=(__bf16)(d.x+ub.x); b4[1]=(__bf16)(d.y+ub.y);
      b4[2]=(__bf16)(d.z+ub.z); b4[3]=(__bf16)(d.w+ub.w);
      *(v4bf*)&Al[row][c4] = b4;
    }
#pragma unroll
    for (int i = 0; i < 4; ++i) {          // B = key/pos rows, already [n][k]
      int f = t + i * 256;
      int n = f >> 3, c4 = (f & 7) << 2;
      const float4 d = *(const float4*)(kp + (size_t)(b*SS + nBase + n)*DM + hoff + k0 + c4);
      v4bf b4; b4[0]=(__bf16)d.x; b4[1]=(__bf16)d.y; b4[2]=(__bf16)d.z; b4[3]=(__bf16)d.w;
      *(v4bf*)&Bl[n][c4] = b4;
    }
    __syncthreads();
#pragma unroll
    for (int mi = 0; mi < 4; ++mi) {
      Frag fa;
      const int ar = wm * 64 + mi * 16 + l;
      fa.h[0] = *(const v8bf*)&Al[ar][half * 8];
      fa.h[1] = *(const v8bf*)&Al[ar][16 + half * 8];
#pragma unroll
      for (int ni = 0; ni < 2; ++ni) {
        Frag fb;
        const int bc = wn * 32 + ni * 16 + l;
        fb.h[0] = *(const v8bf*)&Bl[bc][half * 8];
        fb.h[1] = *(const v8bf*)&Bl[bc][16 + half * 8];
        acc[mi][ni] = wmma_bf16(fa, fb, acc[mi][ni]);
      }
    }
    __syncthreads();
  }
  float* ob = out + (size_t)bh * SS * SS;
#pragma unroll
  for (int mi = 0; mi < 4; ++mi)
#pragma unroll
    for (int ni = 0; ni < 2; ++ni) {
      const int mrow = mBase + wm * 64 + mi * 16 + half * 8;
      const int ncol = nBase + wn * 32 + ni * 16 + l;
#pragma unroll
      for (int r = 0; r < 8; ++r)
        ob[(size_t)(mrow + r) * SS + ncol] = acc[mi][ni][r];
    }
}

// ---------------------------------------------------------------------------
// Rel-shift (as gather) + add + scale + row softmax, in place over content.
// shifted[i,j] = (jsrc==0) ? 0 : pos[isrc, jsrc-1],
//   m = S + i*S + j, isrc = m/(S+1), jsrc = m%(S+1)
// ---------------------------------------------------------------------------
__global__ __launch_bounds__(256) void k_shift_softmax(
    float* __restrict__ attn, const float* __restrict__ pos, float scale)
{
  const int g  = blockIdx.x;
  const int i  = g % SS;
  const int bh = g / SS;
  float* row = attn + (size_t)bh * SS * SS + (size_t)i * SS;
  const float* pb = pos + (size_t)bh * SS * SS;
  __shared__ float red[256];
  const int t = threadIdx.x;

  float s[8];
  float mx = -3.402823466e38f;
#pragma unroll
  for (int e = 0; e < 8; ++e) {
    const int j = e * 256 + t;
    const int m = SS + i * SS + j;
    const int isrc = m / (SS + 1);
    const int jsrc = m - isrc * (SS + 1);
    const float pv = (jsrc == 0) ? 0.f : pb[(size_t)isrc * SS + (jsrc - 1)];
    s[e] = (row[j] + pv) * scale;
    mx = fmaxf(mx, s[e]);
  }
  red[t] = mx; __syncthreads();
  for (int o = 128; o > 0; o >>= 1) { if (t < o) red[t] = fmaxf(red[t], red[t + o]); __syncthreads(); }
  mx = red[0]; __syncthreads();

  float sum = 0.f;
#pragma unroll
  for (int e = 0; e < 8; ++e) { s[e] = __expf(s[e] - mx); sum += s[e]; }
  red[t] = sum; __syncthreads();
  for (int o = 128; o > 0; o >>= 1) { if (t < o) red[t] += red[t + o]; __syncthreads(); }
  const float inv = 1.f / red[0];
#pragma unroll
  for (int e = 0; e < 8; ++e) row[e * 256 + t] = s[e] * inv;
}

// ---------------------------------------------------------------------------
// Context per (b,h): ctx[b,m,h,:] = attn[bh,m,:] @ v[b,:,h,:]  (M=S,N=64,K=S)
// Block tile 128x64, K-step 32; 8 waves as 4(M)x2(N). attn fp32->bf16 staged.
// ---------------------------------------------------------------------------
__global__ __launch_bounds__(256) void k_context(
    const float* __restrict__ attn, const float* __restrict__ vp,
    float* __restrict__ ctx)
{
  __shared__ __bf16 Al[128][48];   // [m][k]
  __shared__ __bf16 Bl[64][48];    // [n=d][k=j]  (v transposed on stage)
  const int bh = blockIdx.y;
  const int b = bh / NH, h = bh % NH, hoff = h * DH;
  const int t = threadIdx.x;
  const int wave = t >> 5, lane = t & 31;
  const int half = lane >> 4, l = lane & 15;
  const int wm = wave >> 1, wn = wave & 1;
  const int mBase = blockIdx.x * 128;
  const float* ab = attn + (size_t)bh * SS * SS;

  v8f acc[2][2];
#pragma unroll
  for (int mi = 0; mi < 2; ++mi)
#pragma unroll
    for (int ni = 0; ni < 2; ++ni) acc[mi][ni] = vzero8();

  for (int j0 = 0; j0 < SS; j0 += 32) {
#pragma unroll
    for (int i = 0; i < 4; ++i) {          // stage attn 128x32
      int f = t + i * 256;
      int row = f >> 3, c4 = (f & 7) << 2;
      const float* ap = ab + (size_t)(mBase + row) * SS + j0 + c4;
      const float4 d = *(const float4*)ap;
      __builtin_prefetch(ap + 32, 0, 3);
      v4bf b4; b4[0]=(__bf16)d.x; b4[1]=(__bf16)d.y; b4[2]=(__bf16)d.z; b4[3]=(__bf16)d.w;
      *(v4bf*)&Al[row][c4] = b4;
    }
#pragma unroll
    for (int i = 0; i < 2; ++i) {          // stage v 32(K)x64(N), transpose
      int f = t + i * 256;
      int kk = f >> 4, n4 = (f & 15) << 2;
      const float4 d = *(const float4*)(vp + (size_t)(b*SS + j0 + kk)*DM + hoff + n4);
      Bl[n4+0][kk]=(__bf16)d.x; Bl[n4+1][kk]=(__bf16)d.y;
      Bl[n4+2][kk]=(__bf16)d.z; Bl[n4+3][kk]=(__bf16)d.w;
    }
    __syncthreads();
#pragma unroll
    for (int mi = 0; mi < 2; ++mi) {
      Frag fa;
      const int ar = wm * 32 + mi * 16 + l;
      fa.h[0] = *(const v8bf*)&Al[ar][half * 8];
      fa.h[1] = *(const v8bf*)&Al[ar][16 + half * 8];
#pragma unroll
      for (int ni = 0; ni < 2; ++ni) {
        Frag fb;
        const int bc = wn * 32 + ni * 16 + l;
        fb.h[0] = *(const v8bf*)&Bl[bc][half * 8];
        fb.h[1] = *(const v8bf*)&Bl[bc][16 + half * 8];
        acc[mi][ni] = wmma_bf16(fa, fb, acc[mi][ni]);
      }
    }
    __syncthreads();
  }
#pragma unroll
  for (int mi = 0; mi < 2; ++mi)
#pragma unroll
    for (int ni = 0; ni < 2; ++ni) {
      const int mrow = mBase + wm * 32 + mi * 16 + half * 8;
      const int ncol = wn * 32 + ni * 16 + l;
#pragma unroll
      for (int r = 0; r < 8; ++r)
        ctx[(size_t)(b*SS + mrow + r) * DM + hoff + ncol] = acc[mi][ni][r];
    }
}

// ---------------------------------------------------------------------------
// Host-side orchestration
// ---------------------------------------------------------------------------
extern "C" void kernel_launch(void* const* d_in, const int* in_sizes, int n_in,
                              void* d_out, int out_size, void* d_ws, size_t ws_size,
                              hipStream_t stream) {
  const float* query = (const float*)d_in[0];
  const float* key   = (const float*)d_in[1];
  const float* value = (const float*)d_in[2];
  const float* pose  = (const float*)d_in[3];
  const float* Wq    = (const float*)d_in[4];
  const float* bq    = (const float*)d_in[5];
  const float* Wk    = (const float*)d_in[6];
  const float* bk    = (const float*)d_in[7];
  const float* Wv    = (const float*)d_in[8];
  const float* bv    = (const float*)d_in[9];
  const float* Wpos  = (const float*)d_in[10];
  const float* ub    = (const float*)d_in[11];
  const float* vb    = (const float*)d_in[12];
  const float* Wout  = (const float*)d_in[13];
  const float* bout  = (const float*)d_in[14];

  float* out  = (float*)d_out;                       // [B,S,D]
  float* attn = out + (size_t)MTOT * DM;             // [B,H,S,S] fp32

  const size_t PROJ = (size_t)MTOT * DM;             // 4096*1024
  float* ws = (float*)d_ws;
  float* q   = ws;
  float* k   = ws + PROJ;
  float* v   = ws + 2 * PROJ;
  float* p   = ws + 3 * PROJ;
  float* ctx = ws + 4 * PROJ;
  float* pscore = ws + 5 * PROJ;                     // [B,H,S,S] fp32

  dim3 gGemm(DM / 128, MTOT / 128);                  // (8, 32)
  k_gemm_bias<<<gGemm, 256, 0, stream>>>(query, Wq, bq,      q, MTOT, DM, DM);
  k_gemm_bias<<<gGemm, 256, 0, stream>>>(key,   Wk, bk,      k, MTOT, DM, DM);
  k_gemm_bias<<<gGemm, 256, 0, stream>>>(value, Wv, bv,      v, MTOT, DM, DM);
  k_gemm_bias<<<gGemm, 256, 0, stream>>>(pose,  Wpos, nullptr, p, MTOT, DM, DM);

  dim3 gScore(SS / 128, SS / 128, BB * NH);          // (16,16,32)
  k_score<<<gScore, 256, 0, stream>>>(q, k, ub, attn);    // content -> d_out attn region
  k_score<<<gScore, 256, 0, stream>>>(q, p, vb, pscore);  // unshifted pos -> ws

  const float scale = 0.03125f;                      // 1/sqrt(1024)
  k_shift_softmax<<<BB * NH * SS, 256, 0, stream>>>(attn, pscore, scale);

  dim3 gCtx(SS / 128, BB * NH);                      // (16, 32)
  k_context<<<gCtx, 256, 0, stream>>>(attn, v, ctx);

  k_gemm_bias<<<gGemm, 256, 0, stream>>>(ctx, Wout, bout, out, MTOT, DM, DM);
}
